// PTTA_78503412236625
// MI455X (gfx1250) — compile-verified, matrix operand order checked by reading, not run on hardware
//
#include <hip/hip_runtime.h>
#include <math.h>

// ---------------- problem constants (from reference setup_inputs) -----------
#define M_ROWS   1024          // feats rows
#define N_BANK   2048          // bank rows
#define D_FEAT   2048          // feature dim
#define N_CLS    1000          // classes
#define IMG_ELEM (3*224*224)   // 150528 floats per image
#define TOPK     8

typedef __bf16 bf16_t;
typedef bf16_t bf16x8  __attribute__((ext_vector_type(8)));
typedef bf16_t bf16x16 __attribute__((ext_vector_type(16)));
typedef float  f32x8   __attribute__((ext_vector_type(8)));
typedef float  f32x4   __attribute__((ext_vector_type(4)));

static __device__ __forceinline__ bf16x16 cat8(bf16x8 a, bf16x8 b) {
    return __builtin_shufflevector(a, b, 0,1,2,3,4,5,6,7,8,9,10,11,12,13,14,15);
}

// ---------------------------------------------------------------------------
// Kernel 1: per-row split fp32 -> (bf16 hi, bf16 lo) and reciprocal L2 norm.
// One 256-thread block per row of 2048 columns (8 columns per thread).
// ---------------------------------------------------------------------------
__global__ __launch_bounds__(256)
void PTTA_prep_kernel(const float* __restrict__ x,
                      bf16_t* __restrict__ hi, bf16_t* __restrict__ lo,
                      float* __restrict__ rnorm) {
    const int row = blockIdx.x;
    const int t   = threadIdx.x;
    const float* xr = x + (size_t)row * D_FEAT;
    bf16_t* hr = hi + (size_t)row * D_FEAT;
    bf16_t* lr = lo + (size_t)row * D_FEAT;

    float ss = 0.0f;
#pragma unroll
    for (int j = 0; j < 8; ++j) {
        const int c = t + j * 256;
        const float v = xr[c];
        ss += v * v;
        const bf16_t h = (bf16_t)v;                 // RNE fp32->bf16
        hr[c] = h;
        lr[c] = (bf16_t)(v - (float)h);             // residual
    }

    __shared__ float red[256];
    red[t] = ss;
    __syncthreads();
#pragma unroll
    for (int s = 128; s > 0; s >>= 1) {
        if (t < s) red[t] += red[t + s];
        __syncthreads();
    }
    if (t == 0) {
        const float n = sqrtf(red[0]);
        rnorm[row] = 1.0f / fmaxf(n, 1e-12f);
    }
}

// ---------------------------------------------------------------------------
// Kernel 2: dist = 1 - (feats @ bank^T) * rf[i] * rb[j]  via split-bf16 WMMA.
// One wave (32 threads) computes one 16x16 output tile; K = 2048 in steps of
// 32, three v_wmma_f32_16x16x32_bf16 per step (hi*hi + hi*lo + lo*hi).
// ---------------------------------------------------------------------------
__global__ __launch_bounds__(32)
void PTTA_gemm_kernel(const bf16_t* __restrict__ Ah, const bf16_t* __restrict__ Al,
                      const bf16_t* __restrict__ Bh, const bf16_t* __restrict__ Bl,
                      const float* __restrict__ rf, const float* __restrict__ rb,
                      float* __restrict__ dist) {
    const int nt   = blockIdx.x;          // N tile (bank)
    const int mt   = blockIdx.y;          // M tile (feats)
    const int lane = threadIdx.x;
    const int half = lane >> 4;           // 0: K 0..7/16..23, 1: K 8..15/24..31
    const int l    = lane & 15;

    const size_t arow = (size_t)(mt * 16 + l) * D_FEAT;   // A row per lane
    const size_t brow = (size_t)(nt * 16 + l) * D_FEAT;   // B^T column = bank row
    const int koff = half * 8;

    f32x8 acc = {};
    for (int kb = 0; kb < D_FEAT; kb += 32) {
        const int k0 = kb + koff;
        // A fragment (16x32 bf16, ISA layout): two 16B chunks per lane
        bf16x16 ah = cat8(*(const bf16x8*)(Ah + arow + k0),
                          *(const bf16x8*)(Ah + arow + k0 + 16));
        bf16x16 al = cat8(*(const bf16x8*)(Al + arow + k0),
                          *(const bf16x8*)(Al + arow + k0 + 16));
        // B fragment (32x16 bf16): mirrored layout, lane = N column
        bf16x16 bh = cat8(*(const bf16x8*)(Bh + brow + k0),
                          *(const bf16x8*)(Bh + brow + k0 + 16));
        bf16x16 bl = cat8(*(const bf16x8*)(Bl + brow + k0),
                          *(const bf16x8*)(Bl + brow + k0 + 16));

        acc = __builtin_amdgcn_wmma_f32_16x16x32_bf16(false, ah, false, bh,
                                                      (short)0, acc, false, false);
        acc = __builtin_amdgcn_wmma_f32_16x16x32_bf16(false, ah, false, bl,
                                                      (short)0, acc, false, false);
        acc = __builtin_amdgcn_wmma_f32_16x16x32_bf16(false, al, false, bh,
                                                      (short)0, acc, false, false);
    }

    // C/D layout: VGPR r -> M = mt*16 + r + 8*half, N = nt*16 + (lane&15)
    const int n = nt * 16 + l;
    const float rbn = rb[n];
#pragma unroll
    for (int r = 0; r < 8; ++r) {
        const int m = mt * 16 + r + half * 8;
        dist[(size_t)m * N_BANK + n] = 1.0f - acc[r] * rf[m] * rbn;
    }
}

// ---------------------------------------------------------------------------
// Kernel 3: exact top-8 per row (jax.lax.top_k semantics: descending value,
// ties -> lower index). 8 sequential argmax passes, 256 threads per row.
// ---------------------------------------------------------------------------
__global__ __launch_bounds__(256)
void PTTA_topk_kernel(const float* __restrict__ dist, int* __restrict__ idxs) {
    __shared__ float sv[256];
    __shared__ int   si[256];
    __shared__ int   chosen[TOPK];
    const int row = blockIdx.x;
    const int t   = threadIdx.x;
    const float* d = dist + (size_t)row * N_BANK;

    float vals[8];
    int   cols[8];
#pragma unroll
    for (int j = 0; j < 8; ++j) {
        const int c = t + j * 256;
        vals[j] = d[c];
        cols[j] = c;
    }

    for (int p = 0; p < TOPK; ++p) {
        float bv = -INFINITY;
        int   bi = 0x7fffffff;
#pragma unroll
        for (int j = 0; j < 8; ++j) {
            bool taken = false;
            for (int q = 0; q < p; ++q)
                if (chosen[q] == cols[j]) taken = true;
            if (!taken && (vals[j] > bv || (vals[j] == bv && cols[j] < bi))) {
                bv = vals[j];
                bi = cols[j];
            }
        }
        sv[t] = bv; si[t] = bi;
        __syncthreads();
#pragma unroll
        for (int s = 128; s > 0; s >>= 1) {
            if (t < s) {
                if (sv[t+s] > sv[t] || (sv[t+s] == sv[t] && si[t+s] < si[t])) {
                    sv[t] = sv[t+s]; si[t] = si[t+s];
                }
            }
            __syncthreads();
        }
        if (t == 0) {
            chosen[p] = si[0];
            idxs[row * TOPK + p] = si[0];
        }
        __syncthreads();
    }
}

// ---------------------------------------------------------------------------
// Kernel 4: grads (mean of bank_features rows), pred_probs (mean of
// bank_probs rows), pred_labels (argmax of pred_probs, ties -> lower index).
// One 256-thread block per output row.
// ---------------------------------------------------------------------------
__global__ __launch_bounds__(256)
void PTTA_gather_kernel(const float* __restrict__ bankf,
                        const float* __restrict__ bankp,
                        const int* __restrict__ idxs,
                        float* __restrict__ out_labels,
                        float* __restrict__ out_probs,
                        float* __restrict__ out_grads) {
    __shared__ int id[TOPK];
    __shared__ float rv[256];
    __shared__ int   rc[256];
    const int i = blockIdx.x;
    const int t = threadIdx.x;
    if (t < TOPK) id[t] = idxs[i * TOPK + t];
    __syncthreads();

    // grads: 2048 columns
#pragma unroll
    for (int j = 0; j < 8; ++j) {
        const int c = t + j * 256;
        float s = 0.0f;
#pragma unroll
        for (int k = 0; k < TOPK; ++k)
            s += bankf[(size_t)id[k] * D_FEAT + c];
        out_grads[(size_t)i * D_FEAT + c] = s * 0.125f;
    }

    // probs + local argmax over 1000 columns
    float bv = -INFINITY;
    int   bc = 0x7fffffff;
#pragma unroll
    for (int j = 0; j < 4; ++j) {
        const int c = t + j * 256;
        if (c < N_CLS) {
            float s = 0.0f;
#pragma unroll
            for (int k = 0; k < TOPK; ++k)
                s += bankp[(size_t)id[k] * N_CLS + c];
            s *= 0.125f;
            out_probs[(size_t)i * N_CLS + c] = s;
            if (s > bv || (s == bv && c < bc)) { bv = s; bc = c; }
        }
    }
    rv[t] = bv; rc[t] = bc;
    __syncthreads();
#pragma unroll
    for (int s = 128; s > 0; s >>= 1) {
        if (t < s) {
            if (rv[t+s] > rv[t] || (rv[t+s] == rv[t] && rc[t+s] < rc[t])) {
                rv[t] = rv[t+s]; rc[t] = rc[t+s];
            }
        }
        __syncthreads();
    }
    if (t == 0) out_labels[i] = (float)rc[0];
}

// ---------------------------------------------------------------------------
// Kernel 5: pred_images = mean of 8 selected images. Bandwidth bound
// (~5.5 GB total traffic -> ~240us at 23.3 TB/s). float4 streaming.
// grid = (150528/4/256 = 147, 1024 rows)
// ---------------------------------------------------------------------------
__global__ __launch_bounds__(256)
void PTTA_images_kernel(const float* __restrict__ images,
                        const int* __restrict__ idxs,
                        float* __restrict__ out_imgs) {
    const int i  = blockIdx.y;
    const int c4 = blockIdx.x * 256 + threadIdx.x;     // 0..37631 (float4 index)

    int id[TOPK];
#pragma unroll
    for (int k = 0; k < TOPK; ++k) id[k] = idxs[i * TOPK + k];

    f32x4 acc = {0.f, 0.f, 0.f, 0.f};
#pragma unroll
    for (int k = 0; k < TOPK; ++k) {
        const f32x4* p = (const f32x4*)(images + (size_t)id[k] * IMG_ELEM);
        acc += p[c4];
    }
    acc *= 0.125f;
    ((f32x4*)(out_imgs + (size_t)i * IMG_ELEM))[c4] = acc;
}

// ---------------------------------------------------------------------------
extern "C" void kernel_launch(void* const* d_in, const int* in_sizes, int n_in,
                              void* d_out, int out_size, void* d_ws, size_t ws_size,
                              hipStream_t stream) {
    const float* feats  = (const float*)d_in[0];   // 1024 x 2048
    const float* bankf  = (const float*)d_in[1];   // 2048 x 2048
    const float* bankp  = (const float*)d_in[2];   // 2048 x 1000
    const float* images = (const float*)d_in[3];   // 2048 x 150528

    // workspace carve (256B aligned)
    char* ws = (char*)d_ws;
    size_t off = 0;
    auto carve = [&](size_t bytes) -> char* {
        char* p = ws + off;
        off += (bytes + 255) & ~(size_t)255;
        return p;
    };
    bf16_t* feats_hi = (bf16_t*)carve((size_t)M_ROWS * D_FEAT * 2);
    bf16_t* feats_lo = (bf16_t*)carve((size_t)M_ROWS * D_FEAT * 2);
    bf16_t* bank_hi  = (bf16_t*)carve((size_t)N_BANK * D_FEAT * 2);
    bf16_t* bank_lo  = (bf16_t*)carve((size_t)N_BANK * D_FEAT * 2);
    float*  rf       = (float*)carve((size_t)M_ROWS * 4);
    float*  rb       = (float*)carve((size_t)N_BANK * 4);
    float*  dist     = (float*)carve((size_t)M_ROWS * N_BANK * 4);
    int*    idxs     = (int*)carve((size_t)M_ROWS * TOPK * 4);

    // output regions, concatenated in reference return order
    float* out        = (float*)d_out;
    float* out_labels = out;                                    // 1024
    float* out_probs  = out + M_ROWS;                           // 1024*1000
    float* out_imgs   = out + M_ROWS + (size_t)M_ROWS * N_CLS;  // 1024*150528
    float* out_grads  = out + M_ROWS + (size_t)M_ROWS * N_CLS
                            + (size_t)M_ROWS * IMG_ELEM;        // 1024*2048

    PTTA_prep_kernel<<<M_ROWS, 256, 0, stream>>>(feats, feats_hi, feats_lo, rf);
    PTTA_prep_kernel<<<N_BANK, 256, 0, stream>>>(bankf, bank_hi, bank_lo, rb);

    PTTA_gemm_kernel<<<dim3(N_BANK / 16, M_ROWS / 16), 32, 0, stream>>>(
        feats_hi, feats_lo, bank_hi, bank_lo, rf, rb, dist);

    PTTA_topk_kernel<<<M_ROWS, 256, 0, stream>>>(dist, idxs);

    PTTA_gather_kernel<<<M_ROWS, 256, 0, stream>>>(bankf, bankp, idxs,
                                                   out_labels, out_probs, out_grads);

    PTTA_images_kernel<<<dim3(IMG_ELEM / 4 / 256, M_ROWS), 256, 0, stream>>>(
        images, idxs, out_imgs);
}
